// torch_RNN_manual_6940667150561
// MI455X (gfx1250) — matrix operand order repaired and checked by reading, not compiled, not run on hardware
//
#include <hip/hip_runtime.h>

// CDNA5 / gfx1250: wave32, WMMA 16x16x32 bf16 with f32 accumulate.
typedef __attribute__((ext_vector_type(16))) __bf16 bf16x16;
typedef __attribute__((ext_vector_type(8)))  __bf16 bf16x8;
typedef __attribute__((ext_vector_type(8)))  float  f32x8;

#define BB 1024
#define TT 2048
#define II 2
#define HH 64
#define WAVES  4    // waves per block
#define NBATCH 16   // batch columns per wave (one N tile)

// Transposed recurrence: h^T = relu( W_h * h^T + W_x * x^T )
//   A (stationary, VGPR-resident): W_h as 4 (M tiles) x 2 (K slabs) bf16 fragments,
//     plus an augmented K-slab holding W_x in K=0..1.
//   B (streaming): h^T read from a per-wave LDS tile, batch = N = lane column;
//     per-lane reads are 32B contiguous (row-major [batch][hidden] bf16 buffer).
//   D: f32 accumulators; packed to bf16 (v_cvt_pk_bf16_f32), ReLU applied on the
//     packed words with v_pk_max_num_bf16 (2 elems/op, inline asm so the backend
//     cannot rewrite it into canonicalize+max), one ds_store_b128 per tile.
__global__ __launch_bounds__(WAVES * 32) void rnn_wmma_kernel(
    const float* __restrict__ x,    // [B, T, 2]
    const float* __restrict__ Wh,   // [64, 64]  (torch Linear layout: [out,in])
    const float* __restrict__ Wx,   // [64, 2]
    float* __restrict__ out)        // [B]  = ||h_T||_2
{
    __shared__ __bf16 lds_h[WAVES * NBATCH * HH];   // 8 KB

    const int tid  = threadIdx.x;
    const int wave = tid >> 5;
    const int lane = tid & 31;
    const int lo   = lane & 15;   // column (batch) within tile / row within A tile
    const int hi   = lane >> 4;   // lane-half selector in WMMA layouts

    __bf16* hbuf = &lds_h[wave * NBATCH * HH];      // private 2 KB slice per wave

    // ---- h_{-1} = 0 ----
    {
        bf16x8 z = {};
        bf16x8* p = (bf16x8*)hbuf;
        #pragma unroll
        for (int i = 0; i < (NBATCH * HH / 8) / 32; ++i)
            p[lane + 32 * i] = z;
    }
    __syncthreads();

    // ---- stationary A fragments: W_h (bf16) ----
    // A k order used here (contiguous, B reads match): lanes hi==0 hold K = 0..15 of the slab,
    // lanes hi==1 hold K = 16..31; M = lo.
    bf16x16 aW[4][2];
    #pragma unroll
    for (int mt = 0; mt < 4; ++mt) {
        #pragma unroll
        for (int kt = 0; kt < 2; ++kt) {
            const float* wrow = Wh + (mt * 16 + lo) * HH + kt * 32 + hi * 16;
            bf16x16 a;
            #pragma unroll
            for (int e = 0; e < 8; ++e) a[e]     = (__bf16)wrow[e];
            #pragma unroll
            for (int e = 0; e < 8; ++e) a[8 + e] = (__bf16)wrow[8 + e];
            aW[mt][kt] = a;
        }
    }
    // augmented K-slab: A = [W_x | 0]  (K=0,1 live in hi==0 lanes, elements 0,1)
    bf16x16 aX[4];
    #pragma unroll
    for (int mt = 0; mt < 4; ++mt) {
        bf16x16 a = {};
        float w0 = Wx[(mt * 16 + lo) * 2 + 0];
        float w1 = Wx[(mt * 16 + lo) * 2 + 1];
        a[0] = (hi == 0) ? (__bf16)w0 : (__bf16)0.f;
        a[1] = (hi == 0) ? (__bf16)w1 : (__bf16)0.f;
        aX[mt] = a;
    }

    const int gw    = blockIdx.x * WAVES + wave;  // global wave id (64 total)
    const int batch = gw * NBATCH + lo;           // this lane's batch column
    const float* xp = x + (size_t)batch * TT * II;

    bf16x16 bx = {};      // augmented B slab: [x^T ; 0] (K=0,1 in hi==0 lanes)

    #pragma unroll 1
    for (int t = 0; t < TT; ++t) {
        // B = h^T from LDS: 16-bit B 32x16 layout -> lane lo is column `batch`,
        // hi selects K half; per-lane 32B contiguous (2x ds_load_b128 each).
        bf16x16 b0 = *(const bf16x16*)&hbuf[lo * HH +  0 + hi * 16];
        bf16x16 b1 = *(const bf16x16*)&hbuf[lo * HH + 32 + hi * 16];

        // x_t injection into augmented B slab
        float2 xv = *(const float2*)(xp + (size_t)t * II);
        bx[0] = (hi == 0) ? (__bf16)xv.x : (__bf16)0.f;
        bx[1] = (hi == 0) ? (__bf16)xv.y : (__bf16)0.f;

        #pragma unroll
        for (int mt = 0; mt < 4; ++mt) {
            f32x8 c = {};
            c = __builtin_amdgcn_wmma_f32_16x16x32_bf16(false, aX[mt],    false, bx,
                                                        (short)0, c, false, false);
            c = __builtin_amdgcn_wmma_f32_16x16x32_bf16(false, aW[mt][0], false, b0,
                                                        (short)0, c, false, false);
            c = __builtin_amdgcn_wmma_f32_16x16x32_bf16(false, aW[mt][1], false, b1,
                                                        (short)0, c, false, false);
            // Pack f32 -> bf16 pairs, then ReLU on packed words:
            // v_pk_max_num_bf16(pk, 0) handles 2 elements per VALU op.
            bf16x8 pk;
            #pragma unroll
            for (int r = 0; r < 8; ++r) pk[r] = (__bf16)c[r];
            uint4 u = __builtin_bit_cast(uint4, pk);
            asm("v_pk_max_num_bf16 %0, %1, 0" : "=v"(u.x) : "v"(u.x));
            asm("v_pk_max_num_bf16 %0, %1, 0" : "=v"(u.y) : "v"(u.y));
            asm("v_pk_max_num_bf16 %0, %1, 0" : "=v"(u.z) : "v"(u.z));
            asm("v_pk_max_num_bf16 %0, %1, 0" : "=v"(u.w) : "v"(u.w));
            // D element r is hidden index m = mt*16 + hi*8 + r at batch column
            // lo: contiguous in the row-major [batch][hidden] buffer.
            *(uint4*)&hbuf[lo * HH + mt * 16 + hi * 8] = u;
        }
    }

    // ---- ||h_T||_2 per batch, from the bf16 state (what the recurrence uses).
    // Lane (lo,hi) reads h[batch=lo][hi*32 .. hi*32+31]; partner lane (^16)
    // holds the other 32 hidden values.
    bf16x16 h0 = *(const bf16x16*)&hbuf[lo * HH + hi * 32];
    bf16x16 h1 = *(const bf16x16*)&hbuf[lo * HH + hi * 32 + 16];
    float p = 0.f;
    #pragma unroll
    for (int e = 0; e < 16; ++e) { float f = (float)h0[e]; p += f * f; }
    #pragma unroll
    for (int e = 0; e < 16; ++e) { float f = (float)h1[e]; p += f * f; }
    p += __shfl_xor(p, 16, 32);
    if (hi == 0) out[batch] = sqrtf(p);
}

extern "C" void kernel_launch(void* const* d_in, const int* in_sizes, int n_in,
                              void* d_out, int out_size, void* d_ws, size_t ws_size,
                              hipStream_t stream) {
    (void)in_sizes; (void)n_in; (void)d_ws; (void)ws_size; (void)out_size;
    const float* x  = (const float*)d_in[0];
    const float* Wh = (const float*)d_in[1];
    const float* Wx = (const float*)d_in[2];
    float* out = (float*)d_out;

    dim3 grid(BB / (WAVES * NBATCH));   // 16 blocks
    dim3 block(WAVES * 32);             // 128 threads = 4 waves (wave32)
    hipLaunchKernelGGL(rnn_wmma_kernel, grid, block, 0, stream, x, Wh, Wx, out);
}
